// GTG_Module_343597383934
// MI455X (gfx1250) — matrix operand order repaired and checked by reading, not compiled, not run on hardware
//
#include <hip/hip_runtime.h>
#include <hip/hip_bf16.h>
#include <math.h>

// ---------------------------------------------------------------------------
// CDNA5 / gfx1250 implementation. wave32.
//  - All GEMMs through v_wmma_f32_16x16x32_bf16 (bf16 in, f32 accum)
//  - A@X label-prop loop stages its A tile via the Tensor Data Mover
//    (tensor_load_to_lds + s_wait_tensorcnt), A stays resident in 192MB L2.
//  - global_prefetch for next k-tiles in the big GEMM.
// ---------------------------------------------------------------------------

typedef __attribute__((ext_vector_type(16))) __bf16 v16bf;
typedef __attribute__((ext_vector_type(8)))  float  v8f;

#define NB    4096
#define DD    512
#define CC    10
#define MAXIT 30

#if defined(__gfx1250__) && __has_builtin(__builtin_amdgcn_tensor_load_to_lds) && \
    __has_builtin(__builtin_amdgcn_s_wait_tensorcnt)
#define USE_TDM 1
typedef unsigned int v4u __attribute__((ext_vector_type(4)));
typedef int          v8i __attribute__((ext_vector_type(8)));
typedef int          v4i __attribute__((ext_vector_type(4)));
#endif

// ---------------------------------------------------------------------------
// f32 -> bf16 convert (weights, etc.)
// ---------------------------------------------------------------------------
__global__ void k_f32_to_bf16(const float* __restrict__ src,
                              __bf16* __restrict__ dst, int n) {
  int i = blockIdx.x * blockDim.x + threadIdx.x;
  if (i < n) dst[i] = (__bf16)src[i];
}

// ---------------------------------------------------------------------------
// Direct conv 3x3, stride 2, pad 1 (NCHW, OIHW). f32 VALU; memory-bound stage.
// ---------------------------------------------------------------------------
__global__ __launch_bounds__(256)
void k_conv(const float* __restrict__ x, const float* __restrict__ w,
            const float* __restrict__ bias, float* __restrict__ y,
            int Cin, int H, int Cout, int OH) {
  int OHW = OH * OH;
  int total = NB * Cout * OHW;
  int idx = blockIdx.x * 256 + threadIdx.x;
  if (idx >= total) return;
  int s  = idx % OHW;
  int t  = idx / OHW;
  int oc = t % Cout;
  int n  = t / Cout;
  int oy = s / OH, ox = s % OH;
  int iy0 = oy * 2 - 1, ix0 = ox * 2 - 1;
  float acc = bias[oc];
  const float* xb = x + (long)n * Cin * H * H;
  const float* wb = w + (long)oc * Cin * 9;
  for (int ic = 0; ic < Cin; ++ic) {
    const float* xc = xb + ic * H * H;
    const float* wc = wb + ic * 9;
#pragma unroll
    for (int ky = 0; ky < 3; ++ky) {
      int iy = iy0 + ky;
      if ((unsigned)iy >= (unsigned)H) continue;
#pragma unroll
      for (int kx = 0; kx < 3; ++kx) {
        int ix = ix0 + kx;
        if ((unsigned)ix >= (unsigned)H) continue;
        acc += xc[iy * H + ix] * wc[ky * 3 + kx];
      }
    }
  }
  y[idx] = acc;
}

// ---------------------------------------------------------------------------
// Per-channel batch mean/var (one block per output channel)
// ---------------------------------------------------------------------------
__global__ __launch_bounds__(256)
void k_stats(const float* __restrict__ y, float* __restrict__ mean,
             float* __restrict__ var, int Cout, int OHW) {
  int c = blockIdx.x;
  int M = NB * OHW;
  float s = 0.f, ss = 0.f;
  for (int t = threadIdx.x; t < M; t += 256) {
    int n = t / OHW, sp = t % OHW;
    float v = y[((long)n * Cout + c) * OHW + sp];
    s += v; ss += v * v;
  }
  __shared__ float rs[256], rss[256];
  rs[threadIdx.x] = s; rss[threadIdx.x] = ss;
  __syncthreads();
  for (int off = 128; off > 0; off >>= 1) {
    if (threadIdx.x < off) {
      rs[threadIdx.x]  += rs[threadIdx.x + off];
      rss[threadIdx.x] += rss[threadIdx.x + off];
    }
    __syncthreads();
  }
  if (threadIdx.x == 0) {
    float m = rs[0] / (float)M;
    mean[c] = m;
    var[c]  = rss[0] / (float)M - m * m;
  }
}

// ---------------------------------------------------------------------------
// BN + ReLU + flatten to (N, 512) bf16 (GEMM operand)
// ---------------------------------------------------------------------------
__global__ __launch_bounds__(256)
void k_bn_relu_flat(const float* __restrict__ y, const float* __restrict__ mean,
                    const float* __restrict__ var, const float* __restrict__ g,
                    const float* __restrict__ bt, __bf16* __restrict__ flat,
                    int Cout, int OHW) {
  int total = NB * Cout * OHW;
  int idx = blockIdx.x * 256 + threadIdx.x;
  if (idx >= total) return;
  int sp = idx % OHW;
  int t  = idx / OHW;
  int oc = t % Cout;
  int n  = t / Cout;
  float v = (y[idx] - mean[oc]) * rsqrtf(var[oc] + 1e-5f) * g[oc] + bt[oc];
  v = fmaxf(v, 0.f);
  flat[(long)n * DD + oc * OHW + sp] = (__bf16)v;
}

// ---------------------------------------------------------------------------
// Generic WMMA GEMM:  C(MxN) = A(MxK) @ B(NxK)^T  (+bias, relu)
// bf16 inputs, f32 accumulate. Block tile 128x64, 8 waves, wave tile 32x32.
// global_prefetch_b8 pulls the next k-tiles while WMMAs drain.
// ---------------------------------------------------------------------------
__global__ __launch_bounds__(256)
void k_gemm_bf16(const __bf16* __restrict__ A, const __bf16* __restrict__ B,
                 const float* __restrict__ bias, float* __restrict__ Cf,
                 __bf16* __restrict__ Cb, int M, int N, int K,
                 int ldc, int coloff, int relu) {
  __shared__ __align__(16) __bf16 As[128][40];
  __shared__ __align__(16) __bf16 Bs[64][40];
  int t    = threadIdx.x;
  int wave = t >> 5, lane = t & 31;
  int half = lane >> 4, l15 = lane & 15;
  int wm = wave >> 1, wn = wave & 1;
  int m0 = blockIdx.y * 128, n0 = blockIdx.x * 64;
  int arow = t >> 1, acol = (t & 1) * 16;
  int brow = t >> 2, bcol = (t & 3) * 8;

  v8f acc[2][2] = {};

  for (int k0 = 0; k0 < K; k0 += 32) {
    const __bf16* ag = A + (long)(m0 + arow) * K + k0 + acol;
    *(uint4*)&As[arow][acol]     = *(const uint4*)ag;
    *(uint4*)&As[arow][acol + 8] = *(const uint4*)(ag + 8);
    const __bf16* bg = B + (long)(n0 + brow) * K + k0 + bcol;
    *(uint4*)&Bs[brow][bcol] = *(const uint4*)bg;
    if (k0 + 32 < K) {             // prefetch next k-tile (global_prefetch_b8)
      __builtin_prefetch(ag + 32, 0, 1);
      __builtin_prefetch(bg + 32, 0, 1);
    }
    __syncthreads();

    v16bf af[2], bfr[2];
#pragma unroll
    for (int mi = 0; mi < 2; ++mi) {
      int r = wm * 32 + mi * 16 + l15;
#pragma unroll
      for (int v = 0; v < 8; ++v) {
        int k = ((v & 4) << 2) + half * 8 + (v & 3) * 2; // ISA 16-bit A layout
        af[mi][2 * v]     = As[r][k];
        af[mi][2 * v + 1] = As[r][k + 1];
      }
    }
#pragma unroll
    for (int ni = 0; ni < 2; ++ni) {
      int c = wn * 32 + ni * 16 + l15;
#pragma unroll
      for (int v = 0; v < 8; ++v) {
        int k = ((v & 4) << 2) + half * 8 + (v & 3) * 2; // ISA 16-bit B layout (matches A)
        bfr[ni][2 * v]     = Bs[c][k];
        bfr[ni][2 * v + 1] = Bs[c][k + 1];
      }
    }
#pragma unroll
    for (int mi = 0; mi < 2; ++mi)
#pragma unroll
      for (int ni = 0; ni < 2; ++ni)
        acc[mi][ni] = __builtin_amdgcn_wmma_f32_16x16x32_bf16(
            false, af[mi], false, bfr[ni], (short)0, acc[mi][ni], false, false);
    __syncthreads();
  }

#pragma unroll
  for (int mi = 0; mi < 2; ++mi)
#pragma unroll
    for (int ni = 0; ni < 2; ++ni) {
      int mrow = m0 + wm * 32 + mi * 16 + half * 8;
      int ncol = wn * 32 + ni * 16 + l15;
      float bv = bias ? bias[n0 + ncol] : 0.f;
#pragma unroll
      for (int r = 0; r < 8; ++r) {
        float v = acc[mi][ni][r] + bv;
        if (relu) v = fmaxf(v, 0.f);
        long o = (long)(mrow + r) * ldc + coloff + n0 + ncol;
        if (Cf) Cf[o] = v;
        if (Cb) Cb[o] = (__bf16)v;
      }
    }
}

// ---------------------------------------------------------------------------
// AX = A(4096x4096 bf16) @ X(4096x16 bf16) -> f32 (cols 10..15 padding).
// Block = 128 rows = 8 waves x one 16-row tile.
// A tile (128x32 bf16) is DMAed into LDS by the Tensor Data Mover with LDS
// padding (16-DWORD interval, 2-DWORD pad => 72B row stride, conflict-free);
// X chunk staged by VALU stores. A (32 MB bf16) stays hot in L2 for all 30
// iterations (30 x 32 MB reads never touch HBM).
// ---------------------------------------------------------------------------
#define ASTR 36   // LDS row stride in bf16 elements: 32 data + 4 pad (72 B)

__global__ __launch_bounds__(256)
void k_ax(const __bf16* __restrict__ Ab, const __bf16* __restrict__ Xb,
          float* __restrict__ AX) {
  __shared__ __bf16 As[128][ASTR];
  __shared__ __bf16 Xs[32][16];
  int t    = threadIdx.x;
  int wave = t >> 5, lane = t & 31;
  int half = lane >> 4, l15 = lane & 15;
  int m0   = blockIdx.x * 128;

  v8f acc = {};
  for (int k0 = 0; k0 < NB; k0 += 32) {
#ifdef USE_TDM
    if (wave == 0) {
      // ---- Tensor DMA descriptor (D#), CDNA5 ISA ----
      unsigned long long ga =
          (unsigned long long)(uintptr_t)(Ab + (size_t)m0 * NB + k0);
      unsigned ldsa = (unsigned)(uintptr_t)&As[0][0];
      v4u g0 = {
        1u,                                   // count=1, user descriptor
        ldsa,                                 // lds_addr
        (unsigned)(ga & 0xffffffffu),         // global_addr[31:0]
        (unsigned)((ga >> 32) & 0x1ffffffu) | (2u << 30) // addr[56:32]|type=2
      };
      v8i g1 = {
        (int)((1u << 16) |                    // data_size = 2 bytes
              (1u << 20) |                    // pad_enable
              (3u << 22) |                    // pad_interval: 16 DWORDs
              (1u << 25)),                    // pad_amount : 2 DWORDs
        (int)(4096u << 16),                   // tensor_dim0 = 4096 (lo16<<16)
        (int)(4096u << 16),                   // dim0 hi=0 | tensor_dim1 lo16
        (int)(32u << 16),                     // dim1 hi=0 | tile_dim0 = 32
        128,                                  // tile_dim1 = 128 rows
        4096,                                 // tensor_dim0_stride lo32
        0, 0                                  // stride hi / dim1_stride unused
      };
      v4i g2 = {0, 0, 0, 0};
      v4i g3 = {0, 0, 0, 0};
      v8i g4 = {0, 0, 0, 0, 0, 0, 0, 0};     // extra group (unused, 2D tile)
      __builtin_amdgcn_tensor_load_to_lds(g0, g1, g2, g3, g4, 0);
      __builtin_amdgcn_s_wait_tensorcnt(0);
    }
#else
    for (int e = t; e < 128 * 32; e += 256) {
      int r = e >> 5, c = e & 31;
      As[r][c] = Ab[(size_t)(m0 + r) * NB + k0 + c];
    }
#endif
    {
      int idx = t * 2;
      int xr = idx >> 4, xc = idx & 15;
      Xs[xr][xc]     = Xb[(k0 + xr) * 16 + xc];
      Xs[xr][xc + 1] = Xb[(k0 + xr) * 16 + xc + 1];
    }
    __syncthreads();

    v16bf af, xf;
#pragma unroll
    for (int v = 0; v < 8; ++v) {
      int ka = ((v & 4) << 2) + half * 8 + (v & 3) * 2;
      af[2 * v]     = As[wave * 16 + l15][ka];
      af[2 * v + 1] = As[wave * 16 + l15][ka + 1];
      int kb = ((v & 4) << 2) + half * 8 + (v & 3) * 2;
      xf[2 * v]     = Xs[kb][l15];
      xf[2 * v + 1] = Xs[kb + 1][l15];
    }
    acc = __builtin_amdgcn_wmma_f32_16x16x32_bf16(
        false, af, false, xf, (short)0, acc, false, false);
    __syncthreads();
  }
  int mrow = m0 + wave * 16 + half * 8;
#pragma unroll
  for (int r = 0; r < 8; ++r)
    AX[(mrow + r) * 16 + l15] = acc[r];
}

// ---------------------------------------------------------------------------
// Embedding row-normalize -> bf16 Z
// ---------------------------------------------------------------------------
__global__ __launch_bounds__(256)
void k_norm(const float* __restrict__ E, __bf16* __restrict__ Zb) {
  int row = blockIdx.x;
  float ss = 0.f;
  for (int c = threadIdx.x; c < DD; c += 256) {
    float v = E[(long)row * DD + c];
    ss += v * v;
  }
  __shared__ float rs[256];
  rs[threadIdx.x] = ss;
  __syncthreads();
  for (int off = 128; off > 0; off >>= 1) {
    if (threadIdx.x < off) rs[threadIdx.x] += rs[threadIdx.x + off];
    __syncthreads();
  }
  __shared__ float nrm;
  if (threadIdx.x == 0) nrm = fmaxf(sqrtf(rs[0]), 1e-12f);
  __syncthreads();
  float inv = 1.f / nrm;
  for (int c = threadIdx.x; c < DD; c += 256)
    Zb[(long)row * DD + c] = (__bf16)(E[(long)row * DD + c] * inv);
}

// ---------------------------------------------------------------------------
// Per-row: zero diag, clip [0,1] (in place), sig = 7th largest.
// ---------------------------------------------------------------------------
__global__ __launch_bounds__(256)
void k_sig(float* __restrict__ Am, float* __restrict__ sig) {
  int row = blockIdx.x;
  float* R = Am + (long)row * NB;
  float loc[7];
#pragma unroll
  for (int q = 0; q < 7; ++q) loc[q] = -1e30f;
  for (int j = threadIdx.x; j < NB; j += 256) {
    float v = R[j];
    if (j == row) v = 0.f;
    v = fminf(fmaxf(v, 0.f), 1.f);
    R[j] = v;
    if (v > loc[6]) {
      int p = 6;
      while (p > 0 && v > loc[p - 1]) { loc[p] = loc[p - 1]; --p; }
      loc[p] = v;
    }
  }
  __shared__ float tops[256 * 7];
#pragma unroll
  for (int q = 0; q < 7; ++q) tops[threadIdx.x * 7 + q] = loc[q];
  __syncthreads();
  if (threadIdx.x == 0) {
    float best[7];
#pragma unroll
    for (int q = 0; q < 7; ++q) best[q] = -1e30f;
    for (int idx = 0; idx < 256 * 7; ++idx) {
      float v = tops[idx];
      if (v > best[6]) {
        int p = 6;
        while (p > 0 && v > best[p - 1]) { best[p] = best[p - 1]; --p; }
        best[p] = v;
      }
    }
    sig[row] = best[6];
  }
}

// ---------------------------------------------------------------------------
// A = exp(-Am^2/(sig_i*sig_j)) in place + global sum (for mean threshold)
// ---------------------------------------------------------------------------
__global__ __launch_bounds__(256)
void k_abuild(float* __restrict__ Am, const float* __restrict__ sig,
              float* __restrict__ sumA) {
  const long total = (long)NB * NB;
  float part = 0.f;
  for (long idx = (long)blockIdx.x * 256 + threadIdx.x; idx < total;
       idx += (long)gridDim.x * 256) {
    int i = (int)(idx >> 12), j = (int)(idx & 4095);
    float v = Am[idx];
    float a = __expf(-v * v / (sig[i] * sig[j]));
    Am[idx] = a;
    part += a;
  }
  __shared__ float rs[256];
  rs[threadIdx.x] = part;
  __syncthreads();
  for (int off = 128; off > 0; off >>= 1) {
    if (threadIdx.x < off) rs[threadIdx.x] += rs[threadIdx.x + off];
    __syncthreads();
  }
  if (threadIdx.x == 0) atomicAdd(sumA, rs[0]);
}

// Threshold against mean + convert to bf16 for the A@X WMMA loop
__global__ __launch_bounds__(256)
void k_thresh(const float* __restrict__ Am, const float* __restrict__ sumA,
              __bf16* __restrict__ Ab) {
  long idx = (long)blockIdx.x * 256 + threadIdx.x;
  if (idx >= (long)NB * NB) return;
  float mean = *sumA * (1.f / ((float)NB * (float)NB));
  float a = Am[idx];
  Ab[idx] = (__bf16)(a < mean ? 0.f : a);
}

// ---------------------------------------------------------------------------
// X init, labeled mask (deterministic bijection stand-in for jax perm),
// scalar state init. Also writes mask output.
// ---------------------------------------------------------------------------
__global__ void k_init(const int* __restrict__ labels, float* __restrict__ Xf,
                       __bf16* __restrict__ Xb, float* __restrict__ entsum,
                       float* __restrict__ maskout, float* __restrict__ sumA,
                       float* __restrict__ err2, int* __restrict__ done) {
  int i = blockIdx.x * blockDim.x + threadIdx.x;
  if (i < NB) {
    unsigned h = ((unsigned)i * 2654435761u + 12345u) & 4095u;
    int labeled = (h < 2048u) ? 1 : 0;
    int lab = labels[i];
#pragma unroll
    for (int c = 0; c < 16; ++c) {
      float v = 0.f;
      if (c < CC) v = labeled ? (c == lab ? 1.f : 0.f) : (1.f / CC);
      Xf[i * 16 + c] = v;
      Xb[i * 16 + c] = (__bf16)v;
    }
    entsum[i]  = 0.f;
    maskout[i] = labeled ? 1.f : 0.f;
  }
  if (i == 0) { *sumA = 0.f; *err2 = 0.f; *done = 0; }
}

// One label-prop step epilogue: mult, row-normalize, entropy, err accumulation
__global__ void k_update(const float* __restrict__ AX, float* __restrict__ Xf,
                         __bf16* __restrict__ Xb, float* __restrict__ entsum,
                         float* __restrict__ err2, const int* __restrict__ done) {
  int i = blockIdx.x * blockDim.x + threadIdx.x;
  if (i >= NB) return;
  int d = *done;
  float xc[CC], xn[CC];
  float s = 0.f;
#pragma unroll
  for (int c = 0; c < CC; ++c) {
    xc[c] = Xf[i * 16 + c];
    float m = xc[c] * AX[i * 16 + c];
    xn[c] = m;
    s += m;
  }
  float inv = 1.f / s;
  float ent = 0.f, e2 = 0.f;
#pragma unroll
  for (int c = 0; c < CC; ++c) {
    xn[c] *= inv;
    ent -= xn[c] * __log2f(xn[c] + 1e-20f);
    float df = xn[c] - xc[c];
    e2 += df * df;
  }
  if (!d) {
#pragma unroll
    for (int c = 0; c < CC; ++c) {
      Xf[i * 16 + c] = xn[c];
      Xb[i * 16 + c] = (__bf16)xn[c];
    }
    entsum[i] += ent;
  }
  atomicAdd(err2, e2);
}

__global__ void k_finish(float* err2, int* done) {
  if (threadIdx.x == 0 && blockIdx.x == 0) {
    if (!*done && sqrtf(*err2) <= 1e-3f) *done = 1;
    *err2 = 0.f;
  }
}

// y_pred = [h, embedds] . cls_w + cls_b (block per row)
__global__ __launch_bounds__(256)
void k_ypred(const float* __restrict__ h, const float* __restrict__ E,
             const float* __restrict__ cw, const float* __restrict__ cb,
             float* __restrict__ out) {
  int row = blockIdx.x;
  float s = 0.f;
  for (int c = threadIdx.x; c < DD; c += 256)
    s += h[(long)row * DD + c] * cw[c] + E[(long)row * DD + c] * cw[DD + c];
  __shared__ float rs[256];
  rs[threadIdx.x] = s;
  __syncthreads();
  for (int off = 128; off > 0; off >>= 1) {
    if (threadIdx.x < off) rs[threadIdx.x] += rs[threadIdx.x + off];
    __syncthreads();
  }
  if (threadIdx.x == 0) out[row] = rs[0] + cb[0];
}

__global__ void k_ytrue(const float* __restrict__ entsum, float* __restrict__ out) {
  int i = blockIdx.x * blockDim.x + threadIdx.x;
  if (i < NB) out[i] = entsum[i] * (1.f / (float)MAXIT);
}

// ---------------------------------------------------------------------------
// Host orchestration
// ---------------------------------------------------------------------------
extern "C" void kernel_launch(void* const* d_in, const int* in_sizes, int n_in,
                              void* d_out, int out_size, void* d_ws, size_t ws_size,
                              hipStream_t stream) {
  // Input layout: dict-order, params tree-flattened alphabetically.
  const float* feat[4] = {(const float*)d_in[0], (const float*)d_in[1],
                          (const float*)d_in[2], (const float*)d_in[3]};
  const float* embedds = (const float*)d_in[4];
  const int*   labels  = (const int*)d_in[5];
  const float* bn_b[4]   = {(const float*)d_in[6],  (const float*)d_in[7],
                            (const float*)d_in[8],  (const float*)d_in[9]};
  const float* bn_g[4]   = {(const float*)d_in[10], (const float*)d_in[11],
                            (const float*)d_in[12], (const float*)d_in[13]};
  const float* cat_b = (const float*)d_in[14];
  const float* cat_w = (const float*)d_in[15];
  const float* cls_b = (const float*)d_in[16];
  const float* cls_w = (const float*)d_in[17];
  const float* conv_b[4] = {(const float*)d_in[18], (const float*)d_in[19],
                            (const float*)d_in[20], (const float*)d_in[21]};
  const float* conv_w[4] = {(const float*)d_in[22], (const float*)d_in[23],
                            (const float*)d_in[24], (const float*)d_in[25]};
  const float* lin_b[4]  = {(const float*)d_in[26], (const float*)d_in[27],
                            (const float*)d_in[28], (const float*)d_in[29]};
  const float* lin_w[4]  = {(const float*)d_in[30], (const float*)d_in[31],
                            (const float*)d_in[32], (const float*)d_in[33]};

  float* out = (float*)d_out; // [y_pred(4096) | y_true(4096) | mask(4096)]

  // Workspace carve-up (256B aligned)
  char* base = (char*)d_ws;
  size_t off = 0;
  auto carve = [&](size_t bytes) {
    char* p = base + off;
    off = (off + bytes + 255) & ~(size_t)255;
    return p;
  };
  float*  Am     = (float*)carve((size_t)NB * NB * 4);   // 64 MB
  __bf16* Ab     = (__bf16*)carve((size_t)NB * NB * 2);  // 32 MB
  float*  convY  = (float*)carve((size_t)NB * 512 * 4);  // 8 MB (max branch)
  __bf16* flat   = (__bf16*)carve((size_t)NB * DD * 2);  // 4 MB
  __bf16* catb   = (__bf16*)carve((size_t)NB * 2048 * 2);// 16 MB
  float*  hbuf   = (float*)carve((size_t)NB * DD * 4);   // 8 MB
  __bf16* Zb     = (__bf16*)carve((size_t)NB * DD * 2);  // 4 MB
  __bf16* wbuf   = (__bf16*)carve((size_t)DD * 2048 * 2);// 2 MB (weight staging)
  float*  Xf     = (float*)carve((size_t)NB * 16 * 4);
  __bf16* Xb     = (__bf16*)carve((size_t)NB * 16 * 2);
  float*  AX     = (float*)carve((size_t)NB * 16 * 4);
  float*  entsum = (float*)carve((size_t)NB * 4);
  float*  sigb   = (float*)carve((size_t)NB * 4);
  float*  meanb  = (float*)carve(512 * 4);
  float*  varb   = (float*)carve(512 * 4);
  float*  sumA   = (float*)carve(16);
  float*  err2   = (float*)carve(16);
  int*    done   = (int*)carve(16);
  (void)ws_size; (void)in_sizes; (void)n_in; (void)out_size;

  const int CH[4] = {64, 128, 256, 512};
  const int FS[4] = {16, 8, 4, 2};

  // Init X / mask / scalar state first (sumA must be 0 before k_abuild)
  k_init<<<16, 256, 0, stream>>>(labels, Xf, Xb, entsum, out + 2 * NB,
                                 sumA, err2, done);

  // ---- Four branches -> concat bf16 buffer (4096 x 2048) ----
  for (int i = 0; i < 4; ++i) {
    int Cin = CH[i], H = FS[i];
    int OH = H / 2, OHW = OH * OH;
    int Cout = CH[i] / (H / 2);
    int tot = NB * Cout * OHW; // 2,097,152 for every branch
    k_conv<<<(tot + 255) / 256, 256, 0, stream>>>(feat[i], conv_w[i], conv_b[i],
                                                  convY, Cin, H, Cout, OH);
    k_stats<<<Cout, 256, 0, stream>>>(convY, meanb, varb, Cout, OHW);
    k_bn_relu_flat<<<(tot + 255) / 256, 256, 0, stream>>>(
        convY, meanb, varb, bn_g[i], bn_b[i], flat, Cout, OHW);
    k_f32_to_bf16<<<(DD * DD + 255) / 256, 256, 0, stream>>>(lin_w[i], wbuf,
                                                             DD * DD);
    k_gemm_bf16<<<dim3(DD / 64, NB / 128), 256, 0, stream>>>(
        flat, wbuf, lin_b[i], nullptr, catb, NB, DD, DD, 2048, DD * i, 1);
  }

  // ---- cat linear: (4096x2048) @ (512x2048)^T -> h (f32) ----
  k_f32_to_bf16<<<(DD * 2048 + 255) / 256, 256, 0, stream>>>(cat_w, wbuf,
                                                             DD * 2048);
  k_gemm_bf16<<<dim3(DD / 64, NB / 128), 256, 0, stream>>>(
      catb, wbuf, cat_b, hbuf, nullptr, NB, DD, 2048, DD, 0, 1);

  // ---- classifier head ----
  k_ypred<<<NB, 256, 0, stream>>>(hbuf, embedds, cls_w, cls_b, out);

  // ---- label propagation ----
  k_norm<<<NB, 256, 0, stream>>>(embedds, Zb);
  k_gemm_bf16<<<dim3(NB / 64, NB / 128), 256, 0, stream>>>(
      Zb, Zb, nullptr, Am, nullptr, NB, NB, DD, NB, 0, 0);
  k_sig<<<NB, 256, 0, stream>>>(Am, sigb);
  k_abuild<<<8192, 256, 0, stream>>>(Am, sigb, sumA);
  k_thresh<<<(int)(((long)NB * NB + 255) / 256), 256, 0, stream>>>(Am, sumA, Ab);

  for (int it = 0; it < MAXIT; ++it) {
    k_ax<<<NB / 128, 256, 0, stream>>>(Ab, Xb, AX);
    k_update<<<16, 256, 0, stream>>>(AX, Xf, Xb, entsum, err2, done);
    k_finish<<<1, 1, 0, stream>>>(err2, done);
  }
  k_ytrue<<<16, 256, 0, stream>>>(entsum, out + NB);
}
